// MultiHeadSelfAttention_1030792151370
// MI455X (gfx1250) — matrix operand order repaired and checked by reading, not compiled, not run on hardware
//
#include <hip/hip_runtime.h>
#include <hip/hip_bf16.h>

// ---------------------------------------------------------------------------
// MHA forward for B=2, N=2048, D=768, H=12, DH=64 on gfx1250 (MI455X).
// Compute-bound (~45 GFLOP over ~40MB of L2-resident data) -> all matmuls via
// v_wmma_f32_16x16x32_bf16. Attention K/V tiles staged to LDS by the Tensor
// Data Mover (tensor_load_to_lds, double-buffered, s_wait_tensorcnt).
// 1/sqrt(DH) folded into Q at QKV-store time.
// ---------------------------------------------------------------------------

typedef __bf16 bf16;
typedef unsigned int u32;
typedef __attribute__((ext_vector_type(16))) __bf16 v16bf;
typedef __attribute__((ext_vector_type(8)))  float  v8f;
typedef __attribute__((ext_vector_type(4)))  unsigned int v4u;
typedef __attribute__((ext_vector_type(8)))  int v8i;
typedef __attribute__((ext_vector_type(4)))  int v4i;

#define BATCH 2
#define SEQ   2048
#define DMODEL 768
#define NHEAD 12
#define DHEAD 64
#define MTOT  (BATCH * SEQ)        // 4096
#define TD    (3 * DMODEL)         // 2304
#define SCALE 0.125f               // 64^-0.5

union FragBF { v16bf v; v4u u[2]; };

__device__ inline v8f zero8() { v8f z = {0.f,0.f,0.f,0.f,0.f,0.f,0.f,0.f}; return z; }

// A-matrix fragment (16 x 32 bf16, row-major src, leading dim ld).
// ISA 16-bit A layout: lane m=lane&15 -> row; per lane two contiguous 16-byte
// runs at k = hi*8 and k = 16 + hi*8.
__device__ inline v16bf load_frag_a(const bf16* __restrict__ src, int ld, int row0, int k0) {
    const int lane = threadIdx.x & 31;
    const int m  = lane & 15;
    const int hi = lane >> 4;
    const bf16* p = src + (size_t)(row0 + m) * ld + k0 + hi * 8;
    FragBF f;
    f.u[0] = *reinterpret_cast<const v4u*>(p);
    f.u[1] = *reinterpret_cast<const v4u*>(p + 16);
    return f.v;
}

// B-matrix fragment from a TRANSPOSED operand: logical B is (32 x 16) but the
// storage is Bt[n][k] row-major with leading dim ld. ISA B layout: lane ->
// column n, two contiguous 16-byte runs at k = hi*8 and k = 16 + hi*8.
__device__ inline v16bf load_frag_bT(const bf16* __restrict__ src, int ld, int k0, int n0) {
    const int lane = threadIdx.x & 31;
    const int n  = n0 + (lane & 15);
    const int hi = lane >> 4;
    const bf16* p = src + (size_t)n * ld + k0 + hi * 8;
    FragBF f;
    f.u[0] = *reinterpret_cast<const v4u*>(p);
    f.u[1] = *reinterpret_cast<const v4u*>(p + 16);
    return f.v;
}

__device__ inline v8f wmma_bf16(v16bf a, v16bf b, v8f c) {
    return __builtin_amdgcn_wmma_f32_16x16x32_bf16(
        false, a, false, b, (short)0, c, false, false);
}

__device__ inline float hmax16(float v) {
    v = fmaxf(v, __shfl_xor(v, 1));
    v = fmaxf(v, __shfl_xor(v, 2));
    v = fmaxf(v, __shfl_xor(v, 4));
    v = fmaxf(v, __shfl_xor(v, 8));
    return v;
}
__device__ inline float hsum16(float v) {
    v += __shfl_xor(v, 1);
    v += __shfl_xor(v, 2);
    v += __shfl_xor(v, 4);
    v += __shfl_xor(v, 8);
    return v;
}

// ---------------- Tensor Data Mover: 2D tile -> LDS (D# per ISA §8) --------
// data_size = 2 bytes. group0: count=1 | lds_addr | global_addr | type=2.
// group1: data_size, tensor_dim0/1, tile_dim0/1, tensor_dim0_stride.
// 6-arg builtin form (clang-23 / therock-10.0): groups 2/3 + trailing group
// zero-filled (2D tile needs only groups 0-1).
__device__ inline void tdm_load_2d(u32 lds_off, const void* gptr,
                                   u32 tile_d0, u32 tile_d1,
                                   u32 tensor_d0, u32 tensor_d1,
                                   u32 stride0_elems) {
    unsigned long long ga = (unsigned long long)(size_t)gptr;
    v4u g0;
    g0[0] = 1u;                                   // count=1 (valid user D#)
    g0[1] = lds_off;                              // lds_addr (bytes)
    g0[2] = (u32)ga;                              // global_addr[31:0]
    g0[3] = (u32)(ga >> 32) | (2u << 30);         // global_addr[56:32] | type=2
    v8i g1;
    g1[0] = (int)(1u << 16);                                      // data_size=2B
    g1[1] = (int)((tensor_d0 & 0xffffu) << 16);                   // tensor_dim0 lo
    g1[2] = (int)((tensor_d0 >> 16) | ((tensor_d1 & 0xffffu) << 16));
    g1[3] = (int)((tensor_d1 >> 16) | (tile_d0 << 16));           // tile_dim0
    g1[4] = (int)tile_d1;                                         // tile_dim1
    g1[5] = (int)stride0_elems;                                   // dim0_stride lo32
    g1[6] = 0;
    g1[7] = 0;
    v4i z4 = {0, 0, 0, 0};
    v8i z8 = {0, 0, 0, 0, 0, 0, 0, 0};
    __builtin_amdgcn_tensor_load_to_lds(g0, g1, z4, z4, z8, 0);
}

__device__ inline u32 lds_offset_of(const void* p) {
    // LDS aperture: low 32 bits of the flat address are the LDS byte offset.
    return (u32)(size_t)p;
}

// --------------------------- conversion kernels ----------------------------

__global__ void cvt_bf16_kernel(const float* __restrict__ src, bf16* __restrict__ dst, int n) {
    int i = blockIdx.x * 256 + threadIdx.x;
    if (i < n) dst[i] = (bf16)src[i];
}

// src: rows x cols (row-major fp32)  ->  dst: cols x rows (row-major bf16)
__global__ void transpose_bf16_kernel(const float* __restrict__ src, bf16* __restrict__ dst,
                                      int rows, int cols) {
    int i = blockIdx.x * 256 + threadIdx.x;
    if (i < rows * cols) {
        int r = i / cols, c = i - r * cols;
        dst[(size_t)c * rows + r] = (bf16)src[i];
    }
}

// ------------------------------- QKV GEMM ----------------------------------
__global__ __launch_bounds__(256) void qkv_kernel(const bf16* __restrict__ xb,
                                                  const bf16* __restrict__ wT,
                                                  const float* __restrict__ bias,
                                                  bf16* __restrict__ q,
                                                  bf16* __restrict__ kmat,
                                                  bf16* __restrict__ vT) {
    const int wave = threadIdx.x >> 5;
    const int lane = threadIdx.x & 31;
    const int row0 = blockIdx.x * 128 + (wave & 1) * 64;   // 4x 16-row tiles
    const int col0 = blockIdx.y * 128 + (wave >> 1) * 32;  // 2x 16-col tiles

    v8f acc[4][2];
    #pragma unroll
    for (int mt = 0; mt < 4; ++mt)
        #pragma unroll
        for (int nt = 0; nt < 2; ++nt) acc[mt][nt] = zero8();

    for (int kk = 0; kk < DMODEL; kk += 32) {
        if (kk + 64 < DMODEL) {  // cover latency for the A/B streams
            __builtin_prefetch(xb + (size_t)(row0 + (lane & 15)) * DMODEL + kk + 64, 0, 0);
            __builtin_prefetch(wT + (size_t)(col0 + (lane & 15)) * DMODEL + kk + 64, 0, 0);
        }
        v16bf a[4], b[2];
        #pragma unroll
        for (int mt = 0; mt < 4; ++mt) a[mt] = load_frag_a(xb, DMODEL, row0 + mt * 16, kk);
        #pragma unroll
        for (int nt = 0; nt < 2; ++nt) b[nt] = load_frag_bT(wT, DMODEL, kk, col0 + nt * 16);
        #pragma unroll
        for (int mt = 0; mt < 4; ++mt)
            #pragma unroll
            for (int nt = 0; nt < 2; ++nt)
                acc[mt][nt] = wmma_bf16(a[mt], b[nt], acc[mt][nt]);
    }

    const int ln = lane & 15, hi = lane >> 4;
    #pragma unroll
    for (int nt = 0; nt < 2; ++nt) {
        const int col = col0 + nt * 16 + ln;
        const float bv = bias[col];
        const int s   = col / DMODEL;           // 0=q 1=k 2=v
        const int rem = col - s * DMODEL;
        const int h = rem >> 6, d = rem & 63;
        #pragma unroll
        for (int mt = 0; mt < 4; ++mt)
            #pragma unroll
            for (int r = 0; r < 8; ++r) {
                const int row = row0 + mt * 16 + r + hi * 8;
                const int bidx = row >> 11, n = row & (SEQ - 1);
                const float v = acc[mt][nt][r] + bv;
                const size_t bh = (size_t)(bidx * NHEAD + h);
                if (s == 0)       q[(bh * SEQ + n) * DHEAD + d]    = (bf16)(v * SCALE);
                else if (s == 1)  kmat[(bh * SEQ + n) * DHEAD + d] = (bf16)v;
                else              vT[(bh * DHEAD + d) * SEQ + n]   = (bf16)v;
            }
    }
}

// --------------------------- flash attention -------------------------------
// grid: (B*H, SEQ/128); 8 waves x 16 queries per workgroup. K/V tiles for the
// whole workgroup are staged into LDS by the TDM (double-buffered), so global
// traffic is issued once per WG instead of once per wave. Online softmax; P is
// transposed from C-layout to A-layout through a wave-private LDS slot.
__global__ __launch_bounds__(256) void attn_kernel(const bf16* __restrict__ q,
                                                   const bf16* __restrict__ kmat,
                                                   const bf16* __restrict__ vT,
                                                   bf16* __restrict__ attn_out) {
    __shared__ __attribute__((aligned(16))) bf16 ldsK[2][32 * DHEAD];  // [key][d]
    __shared__ __attribute__((aligned(16))) bf16 ldsV[2][DHEAD * 32];  // [d][key]
    __shared__ __attribute__((aligned(16))) bf16 lds_p[8][16 * 32];

    const int wave = threadIdx.x >> 5;
    const int lane = threadIdx.x & 31;
    const int ln = lane & 15, hi = lane >> 4;
    const int bh = blockIdx.x;
    const int qbase = blockIdx.y * 128 + wave * 16;

    const bf16* Qp = q    + (size_t)bh * SEQ * DHEAD;
    const bf16* Kp = kmat + (size_t)bh * SEQ * DHEAD;
    const bf16* Vt = vT   + (size_t)bh * DHEAD * SEQ;

    const v16bf a0 = load_frag_a(Qp, DHEAD, qbase, 0);
    const v16bf a1 = load_frag_a(Qp, DHEAD, qbase, 32);

    float m_i[8], l_i[8];
    v8f o[4];
    #pragma unroll
    for (int r = 0; r < 8; ++r) { m_i[r] = -1e30f; l_i[r] = 0.f; }
    #pragma unroll
    for (int t = 0; t < 4; ++t) o[t] = zero8();

    // Prologue: TDM-load K/V tiles for key block 0 into buffer 0.
    if (wave == 0) {
        tdm_load_2d(lds_offset_of(&ldsK[0][0]), Kp, DHEAD, 32, DHEAD, SEQ, DHEAD);
        tdm_load_2d(lds_offset_of(&ldsV[0][0]), Vt, 32, DHEAD, SEQ, DHEAD, SEQ);
    }

    for (int kbi = 0; kbi < SEQ / 32; ++kbi) {
        const int kb  = kbi * 32;
        const int buf = kbi & 1;

        if (wave == 0) __builtin_amdgcn_s_wait_tensorcnt(0);
        __syncthreads();   // tiles[buf] ready; previous reads of tiles[buf^1] done

        // Kick off the DMA for the next key block while we compute this one.
        if (wave == 0 && kbi + 1 < SEQ / 32) {
            const int kn = kb + 32;
            tdm_load_2d(lds_offset_of(&ldsK[buf ^ 1][0]), Kp + (size_t)kn * DHEAD,
                        DHEAD, 32, DHEAD, SEQ, DHEAD);
            tdm_load_2d(lds_offset_of(&ldsV[buf ^ 1][0]), Vt + kn,
                        32, DHEAD, SEQ, DHEAD, SEQ);
        }

        const bf16* Kt = &ldsK[buf][0];   // [key_local][d], ld=64
        const bf16* Vl = &ldsV[buf][0];   // [d][key_local], ld=32

        v8f s0 = zero8(), s1 = zero8();
        s0 = wmma_bf16(a0, load_frag_bT(Kt, DHEAD, 0,  0),  s0);
        s0 = wmma_bf16(a1, load_frag_bT(Kt, DHEAD, 32, 0),  s0);
        s1 = wmma_bf16(a0, load_frag_bT(Kt, DHEAD, 0,  16), s1);
        s1 = wmma_bf16(a1, load_frag_bT(Kt, DHEAD, 32, 16), s1);

        #pragma unroll
        for (int r = 0; r < 8; ++r) {
            const float rmax  = hmax16(fmaxf(s0[r], s1[r]));
            const float mn    = fmaxf(m_i[r], rmax);
            const float alpha = __expf(m_i[r] - mn);
            const float p0 = __expf(s0[r] - mn);
            const float p1 = __expf(s1[r] - mn);
            l_i[r] = l_i[r] * alpha + hsum16(p0 + p1);
            m_i[r] = mn;
            #pragma unroll
            for (int t = 0; t < 4; ++t) o[t][r] *= alpha;
            // C-layout -> LDS (row = M = r + hi*8, col = key-in-block)
            lds_p[wave][(r + hi * 8) * 32 + ln]      = (bf16)p0;
            lds_p[wave][(r + hi * 8) * 32 + ln + 16] = (bf16)p1;
        }

        const v16bf pf = load_frag_a(&lds_p[wave][0], 32, 0, 0);
        #pragma unroll
        for (int t = 0; t < 4; ++t)
            o[t] = wmma_bf16(pf, load_frag_bT(Vl, 32, 0, t * 16), o[t]);

        __syncthreads();   // all waves done with tiles[buf] before next overwrite
    }

    const int b = bh / NHEAD, h = bh - b * NHEAD;
    #pragma unroll
    for (int r = 0; r < 8; ++r) {
        const float inv = 1.0f / l_i[r];
        const int qrow = qbase + r + hi * 8;
        #pragma unroll
        for (int t = 0; t < 4; ++t) {
            const int col = h * DHEAD + t * 16 + ln;
            attn_out[((size_t)b * SEQ + qrow) * DMODEL + col] = (bf16)(o[t][r] * inv);
        }
    }
}

// ----------------------------- output proj ---------------------------------
__global__ __launch_bounds__(256) void proj_kernel(const bf16* __restrict__ A,
                                                   const bf16* __restrict__ wT,
                                                   const float* __restrict__ bias,
                                                   float* __restrict__ out) {
    const int wave = threadIdx.x >> 5;
    const int lane = threadIdx.x & 31;
    const int row0 = blockIdx.x * 128 + (wave & 1) * 64;
    const int col0 = blockIdx.y * 128 + (wave >> 1) * 32;

    v8f acc[4][2];
    #pragma unroll
    for (int mt = 0; mt < 4; ++mt)
        #pragma unroll
        for (int nt = 0; nt < 2; ++nt) acc[mt][nt] = zero8();

    for (int kk = 0; kk < DMODEL; kk += 32) {
        if (kk + 64 < DMODEL) {
            __builtin_prefetch(A  + (size_t)(row0 + (lane & 15)) * DMODEL + kk + 64, 0, 0);
            __builtin_prefetch(wT + (size_t)(col0 + (lane & 15)) * DMODEL + kk + 64, 0, 0);
        }
        v16bf a[4], b[2];
        #pragma unroll
        for (int mt = 0; mt < 4; ++mt) a[mt] = load_frag_a(A, DMODEL, row0 + mt * 16, kk);
        #pragma unroll
        for (int nt = 0; nt < 2; ++nt) b[nt] = load_frag_bT(wT, DMODEL, kk, col0 + nt * 16);
        #pragma unroll
        for (int mt = 0; mt < 4; ++mt)
            #pragma unroll
            for (int nt = 0; nt < 2; ++nt)
                acc[mt][nt] = wmma_bf16(a[mt], b[nt], acc[mt][nt]);
    }

    const int ln = lane & 15, hi = lane >> 4;
    #pragma unroll
    for (int nt = 0; nt < 2; ++nt) {
        const int col = col0 + nt * 16 + ln;
        const float bv = bias[col];
        #pragma unroll
        for (int mt = 0; mt < 4; ++mt)
            #pragma unroll
            for (int r = 0; r < 8; ++r) {
                const int row = row0 + mt * 16 + r + hi * 8;
                out[(size_t)row * DMODEL + col] = acc[mt][nt][r] + bv;
            }
    }
}

// ------------------------------- launcher ----------------------------------
static constexpr size_t ALIGN_UP(size_t x) { return (x + 255) & ~size_t(255); }

extern "C" void kernel_launch(void* const* d_in, const int* in_sizes, int n_in,
                              void* d_out, int out_size, void* d_ws, size_t ws_size,
                              hipStream_t stream) {
    const float* x      = (const float*)d_in[0];
    const float* w_qkv  = (const float*)d_in[1];
    const float* b_qkv  = (const float*)d_in[2];
    const float* w_proj = (const float*)d_in[3];
    const float* b_proj = (const float*)d_in[4];
    float* out = (float*)d_out;

    char* ws = (char*)d_ws;
    size_t off = 0;
    bf16* xb     = (bf16*)(ws + off); off = ALIGN_UP(off + (size_t)MTOT * DMODEL * 2);
    bf16* wqkvT  = (bf16*)(ws + off); off = ALIGN_UP(off + (size_t)TD * DMODEL * 2);
    bf16* wpT    = (bf16*)(ws + off); off = ALIGN_UP(off + (size_t)DMODEL * DMODEL * 2);
    bf16* qm     = (bf16*)(ws + off); off = ALIGN_UP(off + (size_t)BATCH * NHEAD * SEQ * DHEAD * 2);
    bf16* km     = (bf16*)(ws + off); off = ALIGN_UP(off + (size_t)BATCH * NHEAD * SEQ * DHEAD * 2);
    bf16* vTm    = (bf16*)(ws + off); off = ALIGN_UP(off + (size_t)BATCH * NHEAD * SEQ * DHEAD * 2);
    bf16* attn_o = (bf16*)(ws + off); off = ALIGN_UP(off + (size_t)MTOT * DMODEL * 2);
    (void)ws_size; (void)in_sizes; (void)n_in; (void)out_size;

    {
        int n = MTOT * DMODEL;
        cvt_bf16_kernel<<<(n + 255) / 256, 256, 0, stream>>>(x, xb, n);
    }
    {
        int n = DMODEL * TD;
        transpose_bf16_kernel<<<(n + 255) / 256, 256, 0, stream>>>(w_qkv, wqkvT, DMODEL, TD);
    }
    {
        int n = DMODEL * DMODEL;
        transpose_bf16_kernel<<<(n + 255) / 256, 256, 0, stream>>>(w_proj, wpT, DMODEL, DMODEL);
    }

    qkv_kernel<<<dim3(MTOT / 128, TD / 128), 256, 0, stream>>>(xb, wqkvT, b_qkv, qm, km, vTm);
    attn_kernel<<<dim3(BATCH * NHEAD, SEQ / 128), 256, 0, stream>>>(qm, km, vTm, attn_o);
    proj_kernel<<<dim3(MTOT / 128, DMODEL / 128), 256, 0, stream>>>(attn_o, wpT, b_proj, out);
}
